// MeshDeformationBlock_54013508714948
// MI455X (gfx1250) — matrix operand order repaired and checked by reading, not compiled, not run on hardware
//
#include <hip/hip_runtime.h>
#include <hip/hip_bf16.h>
#include <math.h>

// ---------------------------------------------------------------------------
// GATv2 x5 layers on MI455X (gfx1250), wave32 + WMMA.
//
// GEMMs (xl = x@Wl+bl, xr = x@Wr+br) run on v_wmma_f32_16x16x32_f16 with
// 2x4 register blocking (32x64 C macro-tile per wave): 8 WMMAs per K-step on
// 12 global_load_b128 -> ~21 FLOP/B from cache, keeping the matrix pipes fed.
// Fragment layouts per ISA 7.12.2 (f16 A 16x32 / B 32x16 / f32 C 16x16).
//
// Edge phase: destination-CSR built once per call (deterministic: counts +
// block scan + scatter + per-segment sort), then ONE WAVE PER NODE does
// segment-max, exp/denominator and the alpha-weighted gather with plain
// stores -- no float atomics anywhere (determinism + no L2 atomic hotspot).
// ---------------------------------------------------------------------------

typedef _Float16 v16h __attribute__((ext_vector_type(16)));
typedef float    v8f  __attribute__((ext_vector_type(8)));

#define N_NODES 16384
#define N_EDGES 131072
#define ET_EDGES (N_EDGES + N_NODES)   // with self loops
#define NEG_SLOPE 0.2f
#define EPS_DEN 1e-16f

// ---------------- prep: f32 -> padded f16 activations ----------------------
__global__ void k_cvt_x(const float* __restrict__ x, _Float16* __restrict__ xh,
                        int nrows, int fin, int kp) {
  int i = blockIdx.x * blockDim.x + threadIdx.x;
  int total = nrows * kp;
  if (i >= total) return;
  int r = i / kp, k = i - r * kp;
  xh[i] = (k < fin) ? (_Float16)x[(size_t)r * fin + k] : (_Float16)0.f;
}

// W [fin, fout] f32 -> Wt [npad, kp] f16 with Wt[n*kp + k] = W[k*fout + n]
__global__ void k_cvt_w(const float* __restrict__ w, _Float16* __restrict__ wt,
                        int fin, int fout, int kp, int npad) {
  int i = blockIdx.x * blockDim.x + threadIdx.x;
  int total = npad * kp;
  if (i >= total) return;
  int n = i / kp, k = i - n * kp;
  wt[i] = (n < fout && k < fin) ? (_Float16)w[(size_t)k * fout + n] : (_Float16)0.f;
}

// ---------------- WMMA GEMM, 2x4 blocked (NcPad % 64 == 0, M % 32 == 0) ----
__global__ __launch_bounds__(128) void
k_gemm_wmma_2x4(const _Float16* __restrict__ A, const _Float16* __restrict__ Bt,
                const float* __restrict__ bias, float* __restrict__ C,
                int M, int Kp, int NcPad, int NcReal) {
  const int lane = threadIdx.x & 31;
  const int wave = threadIdx.x >> 5;
  const int tilesN = NcPad >> 6;               // 64-wide macro-columns
  const int tile = blockIdx.x * (blockDim.x >> 5) + wave;
  const int totalTiles = (M >> 5) * tilesN;
  if (tile >= totalTiles) return;              // wave-uniform -> EXEC all-1s
  const int tm  = (tile / tilesN) << 5;
  const int tn  = (tile % tilesN) << 6;
  const int l16 = lane & 15;
  const int hi  = lane >> 4;

  const _Float16* Ap0 = A + (size_t)(tm + l16) * Kp + hi * 8;
  const _Float16* Ap1 = Ap0 + (size_t)16 * Kp;
  const _Float16* Bp  = Bt + (size_t)(tn + l16) * Kp + hi * 8;
  const size_t bstride = (size_t)16 * Kp;      // 16 Bt rows between j-tiles

  v8f acc[2][4] = {};
  for (int kb = 0; kb < Kp; kb += 32) {
    union U { v16h v; float4 f[2]; };
    U a[2], b[4];
    a[0].f[0] = *(const float4*)(Ap0 + kb);
    a[0].f[1] = *(const float4*)(Ap0 + kb + 16);
    a[1].f[0] = *(const float4*)(Ap1 + kb);
    a[1].f[1] = *(const float4*)(Ap1 + kb + 16);
#pragma unroll
    for (int j = 0; j < 4; ++j) {
      const _Float16* bp = Bp + (size_t)j * bstride + kb;
      b[j].f[0] = *(const float4*)(bp);
      b[j].f[1] = *(const float4*)(bp + 16);
    }
#pragma unroll
    for (int i = 0; i < 2; ++i)
#pragma unroll
      for (int j = 0; j < 4; ++j)
        acc[i][j] = __builtin_amdgcn_wmma_f32_16x16x32_f16(
            false, a[i].v, false, b[j].v, (short)0, acc[i][j], false, false);
  }

#pragma unroll
  for (int i = 0; i < 2; ++i)
#pragma unroll
    for (int j = 0; j < 4; ++j) {
      const int bcol = tn + j * 16 + l16;
      const float bv = (bcol < NcReal) ? bias[bcol] : 0.f;
      float* Cp = C + (size_t)(tm + i * 16 + hi * 8) * NcPad + bcol;
#pragma unroll
      for (int v = 0; v < 8; ++v)
        Cp[(size_t)v * NcPad] = acc[i][j][v] + bv;
    }
}

// ---------------- WMMA GEMM, simple 1-tile (for ragged small N) ------------
__global__ void k_gemm_wmma(const _Float16* __restrict__ A,
                            const _Float16* __restrict__ Bt,
                            const float* __restrict__ bias,
                            float* __restrict__ C,
                            int M, int Kp, int NcPad, int NcReal) {
  const int lane = threadIdx.x & 31;
  const int wave = threadIdx.x >> 5;
  const int ntn  = NcPad >> 4;
  const int tile = blockIdx.x * (blockDim.x >> 5) + wave;
  const int totalTiles = (M >> 4) * ntn;
  if (tile >= totalTiles) return;
  const int tm  = (tile / ntn) << 4;
  const int tn  = (tile % ntn) << 4;
  const int l16 = lane & 15;
  const int hi  = lane >> 4;

  const _Float16* Ap = A  + (size_t)(tm + l16) * Kp + hi * 8;
  const _Float16* Bp = Bt + (size_t)(tn + l16) * Kp + hi * 8;

  v8f acc = {};
  for (int kb = 0; kb < Kp; kb += 32) {
    union { v16h v; float4 f[2]; } a, b;
    a.f[0] = *(const float4*)(Ap + kb);
    a.f[1] = *(const float4*)(Ap + kb + 16);
    b.f[0] = *(const float4*)(Bp + kb);
    b.f[1] = *(const float4*)(Bp + kb + 16);
    acc = __builtin_amdgcn_wmma_f32_16x16x32_f16(
        false, a.v, false, b.v, (short)0, acc, false, false);
  }
  const int  bcol = tn + l16;
  const float bv  = (bcol < NcReal) ? bias[bcol] : 0.f;
  float* Cp = C + (size_t)(tm + hi * 8) * NcPad + bcol;
#pragma unroll
  for (int v = 0; v < 8; ++v)
    Cp[(size_t)v * NcPad] = acc[v] + bv;
}

// ---------------- CSR-by-destination build (deterministic) -----------------
__device__ __forceinline__ int edge_dst(const int* edst, int e) {
  return (e < N_EDGES) ? edst[e] : (e - N_EDGES);
}
__device__ __forceinline__ int edge_src(const int* esrc, int e) {
  return (e < N_EDGES) ? esrc[e] : (e - N_EDGES);
}

__global__ void k_zero_i32(int* __restrict__ p, int n) {
  int i = blockIdx.x * blockDim.x + threadIdx.x;
  if (i < n) p[i] = 0;
}
__global__ void k_copy_i32(int* __restrict__ dst, const int* __restrict__ src, int n) {
  int i = blockIdx.x * blockDim.x + threadIdx.x;
  if (i < n) dst[i] = src[i];
}
__global__ void k_count(const int* __restrict__ edst, int* __restrict__ cnt) {
  int e = blockIdx.x * blockDim.x + threadIdx.x;
  if (e >= ET_EDGES) return;
  atomicAdd(cnt + edge_dst(edst, e), 1);
}

#define SCAN_T 1024
__global__ void k_scan(const int* __restrict__ cnt, int* __restrict__ off, int n) {
  __shared__ int part[SCAN_T];
  const int t = threadIdx.x;
  const int per = n / SCAN_T;
  const int base = t * per;
  int s = 0;
  for (int i = 0; i < per; ++i) s += cnt[base + i];
  part[t] = s;
  __syncthreads();
  for (int d = 1; d < SCAN_T; d <<= 1) {
    int v = (t >= d) ? part[t - d] : 0;
    __syncthreads();
    part[t] += v;
    __syncthreads();
  }
  int run = (t == 0) ? 0 : part[t - 1];
  for (int i = 0; i < per; ++i) {
    off[base + i] = run;
    run += cnt[base + i];
  }
  if (t == SCAN_T - 1) off[n] = run;
}

__global__ void k_scatter(const int* __restrict__ edst, int* __restrict__ cursor,
                          int* __restrict__ eid) {
  int e = blockIdx.x * blockDim.x + threadIdx.x;
  if (e >= ET_EDGES) return;
  int pos = atomicAdd(cursor + edge_dst(edst, e), 1);
  eid[pos] = e;
}

// per-node insertion sort of its edge ids -> deterministic summation order
__global__ void k_sort_seg(int* __restrict__ eid, const int* __restrict__ rowoff) {
  int d = blockIdx.x * blockDim.x + threadIdx.x;
  if (d >= N_NODES) return;
  int b = rowoff[d], e = rowoff[d + 1];
  for (int i = b + 1; i < e; ++i) {
    int v = eid[i], j = i - 1;
    while (j >= b && eid[j] > v) { eid[j + 1] = eid[j]; --j; }
    eid[j + 1] = v;
  }
}

// ---------------- edge scores (wave per edge) ------------------------------
__global__ void k_edge_score(const float* __restrict__ xl, const float* __restrict__ xr,
                             const float* __restrict__ att,
                             const int* __restrict__ esrc, const int* __restrict__ edst,
                             float* __restrict__ escore, int F, int ld) {
  int gid  = blockIdx.x * blockDim.x + threadIdx.x;
  int e    = gid >> 5;
  int lane = gid & 31;
  if (e >= ET_EDGES) return;
  int s = edge_src(esrc, e);
  int d = edge_dst(edst, e);
  const float* pl = xl + (size_t)s * ld;
  const float* pr = xr + (size_t)d * ld;
  float acc = 0.f;
  for (int f = lane; f < F; f += 32) {
    float h = pl[f] + pr[f];
    h = (h > 0.f) ? h : NEG_SLOPE * h;
    acc += h * att[f];
  }
#pragma unroll
  for (int off = 16; off > 0; off >>= 1)
    acc += __shfl_down(acc, off, 32);
  if (lane == 0) escore[e] = acc;
}

// ---------------- per-node softmax + aggregate (wave per node) -------------
__global__ void k_node_agg(const float* __restrict__ xl, const float* __restrict__ escore,
                           const int* __restrict__ esrc, const int* __restrict__ eid,
                           const int* __restrict__ rowoff, const float* __restrict__ bias,
                           float* __restrict__ wexp, float* __restrict__ out,
                           int F, int ldx, int ldo, int relu) {
  int gid  = blockIdx.x * blockDim.x + threadIdx.x;
  int d    = gid >> 5;
  int lane = gid & 31;
  if (d >= N_NODES) return;
  const int begin = rowoff[d], end = rowoff[d + 1];

  // segment max
  float m = -INFINITY;
  for (int t = begin + lane; t < end; t += 32) m = fmaxf(m, escore[eid[t]]);
#pragma unroll
  for (int o = 16; o > 0; o >>= 1) m = fmaxf(m, __shfl_xor(m, o, 32));
  if (!isfinite(m)) m = 0.f;

  // exp weights + denominator
  float ssum = 0.f;
  for (int t = begin + lane; t < end; t += 32) {
    int e = eid[t];
    float w = expf(escore[e] - m);
    wexp[e] = w;
    ssum += w;
  }
#pragma unroll
  for (int o = 16; o > 0; o >>= 1) ssum += __shfl_xor(ssum, o, 32);
  const float inv = 1.f / (ssum + EPS_DEN);

  // alpha-weighted gather of source features (deterministic order t asc)
  for (int f = lane; f < F; f += 32) {
    float acc = 0.f;
    for (int t = begin; t < end; ++t) {
      int e = eid[t];
      int s = edge_src(esrc, e);
      acc += wexp[e] * xl[(size_t)s * ldx + f];
    }
    float v = acc * inv + bias[f];
    if (relu) v = fmaxf(v, 0.f);
    out[(size_t)d * ldo + f] = v;
  }
}

__global__ void k_copy(float* __restrict__ dst, const float* __restrict__ src, int n) {
  int i = blockIdx.x * blockDim.x + threadIdx.x;
  if (i < n) dst[i] = src[i];
}

// ---------------------------------------------------------------------------
extern "C" void kernel_launch(void* const* d_in, const int* in_sizes, int n_in,
                              void* d_out, int out_size, void* d_ws, size_t ws_size,
                              hipStream_t stream) {
  (void)in_sizes; (void)n_in; (void)out_size; (void)ws_size;

  static const int FIN[5]  = {2307, 1024, 512, 256, 128};
  static const int FOUT[5] = {1024,  512, 256, 128,   3};

  const float* x_in = (const float*)d_in[0];
  const int*   ei   = (const int*)d_in[31];     // edge_index [2, E]
  const int*   esrc = ei;
  const int*   edst = ei + N_EDGES;

  // ---- carve workspace (256B aligned chunks) ----
  char* wsp = (char*)d_ws;
  auto take = [&](size_t bytes) {
    char* p = wsp;
    wsp += (bytes + 255) & ~(size_t)255;
    return p;
  };
  _Float16* xh  = (_Float16*)take((size_t)N_NODES * 2336 * 2);
  _Float16* wlt = (_Float16*)take((size_t)1024 * 2336 * 2);
  _Float16* wrt = (_Float16*)take((size_t)1024 * 2336 * 2);
  float* xl     = (float*)take((size_t)N_NODES * 1024 * 4);
  float* xr     = (float*)take((size_t)N_NODES * 1024 * 4);
  float* act    = (float*)take((size_t)N_NODES * 1024 * 4);
  float* escore = (float*)take((size_t)ET_EDGES * 4);
  float* wexp   = (float*)take((size_t)ET_EDGES * 4);
  int* cnt      = (int*)take((size_t)N_NODES * 4);
  int* rowoff   = (int*)take((size_t)(N_NODES + 1) * 4);
  int* cursor   = (int*)take((size_t)N_NODES * 4);
  int* eid      = (int*)take((size_t)ET_EDGES * 4);

  const int TPB = 256;
  auto cdiv = [](int a, int b) { return (a + b - 1) / b; };

  // ---- CSR-by-destination (deterministic, rebuilt every call) ----
  k_zero_i32<<<cdiv(N_NODES, TPB), TPB, 0, stream>>>(cnt, N_NODES);
  k_count<<<cdiv(ET_EDGES, TPB), TPB, 0, stream>>>(edst, cnt);
  k_scan<<<1, SCAN_T, 0, stream>>>(cnt, rowoff, N_NODES);
  k_copy_i32<<<cdiv(N_NODES, TPB), TPB, 0, stream>>>(cursor, rowoff, N_NODES);
  k_scatter<<<cdiv(ET_EDGES, TPB), TPB, 0, stream>>>(edst, cursor, eid);
  k_sort_seg<<<cdiv(N_NODES, TPB), TPB, 0, stream>>>(eid, rowoff);

  for (int l = 0; l < 5; ++l) {
    const int fin  = FIN[l],  fout = FOUT[l];
    const int kp   = ((fin + 31) / 32) * 32;     // K padded to 32
    const int npad = ((fout + 15) / 16) * 16;    // N padded to 16

    const float* Wl   = (const float*)d_in[1 + 6 * l + 0];
    const float* bl   = (const float*)d_in[1 + 6 * l + 1];
    const float* Wr   = (const float*)d_in[1 + 6 * l + 2];
    const float* br   = (const float*)d_in[1 + 6 * l + 3];
    const float* att  = (const float*)d_in[1 + 6 * l + 4];
    const float* bias = (const float*)d_in[1 + 6 * l + 5];

    const float* Xin = (l == 0) ? x_in : act;
    float* outp = (l < 4) ? act : ((float*)d_out + (size_t)N_NODES * 128);

    // f32 -> padded f16
    k_cvt_x<<<cdiv(N_NODES * kp, TPB), TPB, 0, stream>>>(Xin, xh, N_NODES, fin, kp);
    k_cvt_w<<<cdiv(npad * kp, TPB), TPB, 0, stream>>>(Wl, wlt, fin, fout, kp, npad);
    k_cvt_w<<<cdiv(npad * kp, TPB), TPB, 0, stream>>>(Wr, wrt, fin, fout, kp, npad);

    // two WMMA GEMMs
    if ((npad & 63) == 0) {
      const int tiles  = (N_NODES / 32) * (npad / 64);
      const int blocks = cdiv(tiles, 4);
      k_gemm_wmma_2x4<<<blocks, 128, 0, stream>>>(xh, wlt, bl, xl, N_NODES, kp, npad, fout);
      k_gemm_wmma_2x4<<<blocks, 128, 0, stream>>>(xh, wrt, br, xr, N_NODES, kp, npad, fout);
    } else {
      const int tiles  = (N_NODES / 16) * (npad / 16);
      const int blocks = cdiv(tiles, 4);
      k_gemm_wmma<<<blocks, 128, 0, stream>>>(xh, wlt, bl, xl, N_NODES, kp, npad, fout);
      k_gemm_wmma<<<blocks, 128, 0, stream>>>(xh, wrt, br, xr, N_NODES, kp, npad, fout);
    }

    // edge scores, then per-node softmax + aggregation (bias/relu fused)
    k_edge_score<<<ET_EDGES / 8, TPB, 0, stream>>>(xl, xr, att, esrc, edst,
                                                   escore, fout, npad);
    k_node_agg<<<(N_NODES * 32) / TPB, TPB, 0, stream>>>(
        xl, escore, esrc, eid, rowoff, bias, wexp, outp,
        fout, npad, fout, (l < 4) ? 1 : 0);

    // first tuple element: layer-3 activations [N,128]
    if (l == 3)
      k_copy<<<cdiv(N_NODES * 128, TPB), TPB, 0, stream>>>((float*)d_out, act,
                                                           N_NODES * 128);
  }
}
